// LinearAttention_39676907888607
// MI455X (gfx1250) — matrix operand order repaired and checked by reading, not compiled, not run on hardware
//
#include <hip/hip_runtime.h>
#include <stdint.h>

// ---------------------------------------------------------------------------
// LinearAttention fused pipeline for MI455X (gfx1250, wave32, WMMA).
//  b=8, dim=512, n=8192, heads=8, dim_head=32, hidden=256.
//  ~71 GFLOP, dominated by two dense GEMMs -> bf16 v_wmma_f32_16x16x32_bf16.
//  LayerNorm folded into QKV GEMM epilogue; x transposed once to a
//  K-contiguous bf16 buffer.  GEMM B-tiles staged in LDS with CDNA5
//  global_load_async_to_lds_b128 (ASYNCcnt) + double buffering so the DMA of
//  tile k+1 overlaps the WMMAs of tile k and all 8 waves share one copy.
// ---------------------------------------------------------------------------

#define B_   8
#define C_   512
#define N_   8192
#define HID  256
#define O3   768
#define H_   8
#define DH   32
#define BN   (B_ * N_)
#define SCALE_ 0.17677669529663687f
#define EPS_ 1e-5f

typedef __attribute__((ext_vector_type(16))) __bf16 v16bf;
typedef __attribute__((ext_vector_type(8)))  float  v8f;

__device__ __forceinline__ unsigned short f2bf(float f) {
  union { float f; unsigned u; } v; v.f = f;
  unsigned r = v.u + 0x7FFFu + ((v.u >> 16) & 1u);   // round-to-nearest-even
  return (unsigned short)(r >> 16);
}
__device__ __forceinline__ float bf2f(unsigned short b) {
  union { unsigned u; float f; } v; v.u = ((unsigned)b) << 16;
  return v.f;
}

union FragBF { v16bf v; uint4 q[2]; unsigned short s[16]; };

// A-matrix 16x32 bf16 fragment per CDNA5 ISA layout:
// lanes 0-15: row M=lane, K = {0..7} and {16..23}; lanes 16-31: row M=lane-16,
// K = {8..15} and {24..31}.  base points at row 0 of the tile.
__device__ __forceinline__ v16bf load_frag_a(const unsigned short* __restrict__ base,
                                             int ld, int k0) {
  int lane = threadIdx.x & 31;
  int row  = lane & 15;
  int grp  = lane >> 4;
  const unsigned short* p = base + (size_t)row * ld + k0 + grp * 8;
  FragBF u;
  u.q[0] = *reinterpret_cast<const uint4*>(p);        // K 0..7  or 8..15
  u.q[1] = *reinterpret_cast<const uint4*>(p + 16);   // K 16..23 or 24..31
  return u.v;
}

// B-matrix 32x16 bf16 fragment: lane = column (mod 16); lanes 0-15 hold
// K={0..7,16..23}, lanes 16-31 hold K={8..15,24..31}.  Source stored
// column-major-over-K, base at column 0 of tile.
__device__ __forceinline__ v16bf load_frag_b(const unsigned short* __restrict__ base,
                                             int ld, int k0) {
  int lane = threadIdx.x & 31;
  int col  = lane & 15;
  int grp  = lane >> 4;
  const unsigned short* p = base + (size_t)col * ld + k0 + grp * 8;
  FragBF u;
  u.q[0] = *reinterpret_cast<const uint4*>(p);
  u.q[1] = *reinterpret_cast<const uint4*>(p + 16);
  return u.v;
}

// B fragment from an LDS-staged [cols][32] bf16 tile (ds_load_b128 x2).
__device__ __forceinline__ v16bf load_frag_b_lds(const unsigned short* base) {
  int lane = threadIdx.x & 31;
  int col  = lane & 15;
  int grp  = lane >> 4;
  const unsigned short* p = base + col * 32 + grp * 8;
  FragBF u;
  u.q[0] = *reinterpret_cast<const uint4*>(p);
  u.q[1] = *reinterpret_cast<const uint4*>(p + 16);
  return u.v;
}

__device__ __forceinline__ v8f wmma_bf16(v16bf a, v16bf b, v8f c) {
  return __builtin_amdgcn_wmma_f32_16x16x32_bf16(false, a, false, b,
                                                 (short)0, c, false, false);
}

// CDNA5 async copy: global -> LDS, 16B per lane, tracked by ASYNCcnt.
// lds_off is the 32-bit LDS byte offset (low 32 bits of the flat address).
__device__ __forceinline__ void async_g2l_b128(unsigned lds_off,
                                               const void* gaddr) {
  asm volatile("global_load_async_to_lds_b128 %0, %1, off"
               :: "v"(lds_off), "v"(gaddr) : "memory");
}
__device__ __forceinline__ void wait_asynccnt0() {
  asm volatile("s_wait_asynccnt 0x0" ::: "memory");
}

// ---------------------------------------------------------------------------
// K0: weight prep.  blocks [0,768): wqg[o][c]=bf16(w_qkv*g), wsum[o], wb[o].
//     blocks [768,1280): wout_bf16.
// ---------------------------------------------------------------------------
__global__ void k0_weights(const float* __restrict__ w_qkv,
                           const float* __restrict__ g,
                           const float* __restrict__ b_ln,
                           const float* __restrict__ w_out,
                           unsigned short* __restrict__ wqg,
                           unsigned short* __restrict__ wout_bf,
                           float* __restrict__ wsum,
                           float* __restrict__ wb) {
  int blk = blockIdx.x, tid = threadIdx.x;
  if (blk < O3) {
    __shared__ float r1[256], r2[256];
    float s1 = 0.f, s2 = 0.f;
    for (int c = tid; c < C_; c += 256) {
      float w  = w_qkv[(size_t)blk * C_ + c];
      float wg = w * g[c];
      wqg[(size_t)blk * C_ + c] = f2bf(wg);
      s1 += wg;
      s2 += w * b_ln[c];
    }
    r1[tid] = s1; r2[tid] = s2; __syncthreads();
    for (int s = 128; s > 0; s >>= 1) {
      if (tid < s) { r1[tid] += r1[tid + s]; r2[tid] += r2[tid + s]; }
      __syncthreads();
    }
    if (tid == 0) { wsum[blk] = r1[0]; wb[blk] = r2[0]; }
  } else {
    int o = blk - O3;                    // 0..511
    int c = tid;                         // 0..255 == HID
    wout_bf[(size_t)o * HID + c] = f2bf(w_out[(size_t)o * HID + c]);
  }
}

// ---------------------------------------------------------------------------
// K1: LayerNorm stats + transpose x -> xt[j][c] bf16 (j = b*N+n, c contiguous).
// ---------------------------------------------------------------------------
__global__ void k1_ln_transpose(const float* __restrict__ x,
                                unsigned short* __restrict__ xt,
                                float* __restrict__ rstd,
                                float* __restrict__ tmu) {
  __shared__ float tile[16][65];
  __shared__ float r1[256], r2[256];
  int tid = threadIdx.x;
  int b   = blockIdx.x >> 7;            // /128
  int n0  = (blockIdx.x & 127) * 64;
  int nl  = tid & 63;
  float sum = 0.f, sq = 0.f;

  struct alignas(8) US4 { unsigned short a, b, c, d; };

  for (int c0 = 0; c0 < C_; c0 += 16) {
    for (int it = 0; it < 4; ++it) {
      int cl = it * 4 + (tid >> 6);
      float v = x[((size_t)b * C_ + c0 + cl) * N_ + n0 + nl];
      tile[cl][nl] = v;
      sum += v; sq += v * v;
    }
    __syncthreads();
    int jl = tid >> 2, cq = tid & 3;
    US4 p;
    p.a = f2bf(tile[cq * 4 + 0][jl]);
    p.b = f2bf(tile[cq * 4 + 1][jl]);
    p.c = f2bf(tile[cq * 4 + 2][jl]);
    p.d = f2bf(tile[cq * 4 + 3][jl]);
    *reinterpret_cast<US4*>(xt + (size_t)(b * N_ + n0 + jl) * C_ + c0 + cq * 4) = p;
    __syncthreads();
  }
  r1[tid] = sum; r2[tid] = sq; __syncthreads();
  if (tid < 64) {
    float s  = r1[tid] + r1[tid + 64] + r1[tid + 128] + r1[tid + 192];
    float s2 = r2[tid] + r2[tid + 64] + r2[tid + 128] + r2[tid + 192];
    float mean = s * (1.0f / C_);
    float var  = s2 * (1.0f / C_) - mean * mean;     // biased, matches jnp.var
    float rs   = rsqrtf(var + EPS_);
    int j = b * N_ + n0 + tid;
    rstd[j] = rs;
    tmu[j]  = mean * rs;
  }
}

// ---------------------------------------------------------------------------
// K2: QKV GEMM  G[768, 65536] = Wg(bf16) @ Xt(bf16), LN folded into epilogue.
//     B-tile (128 cols x 32 K, 8KB) async-staged to LDS, double buffered;
//     one copy shared by all 8 waves.  Wave owns one 16-row m-tile x 128 cols.
// ---------------------------------------------------------------------------
__global__ void k2_qkv_gemm(const unsigned short* __restrict__ wqg,
                            const unsigned short* __restrict__ xt,
                            const float* __restrict__ rstd,
                            const float* __restrict__ tmu,
                            const float* __restrict__ wsum,
                            const float* __restrict__ wb,
                            unsigned short* __restrict__ qb,
                            unsigned short* __restrict__ kb,
                            unsigned short* __restrict__ vb) {
  __shared__ __align__(16) unsigned short tileB[2][128][32];
  int tid = threadIdx.x;
  int wave = tid >> 5, lane = tid & 31;
  int grp = lane >> 4, nn = lane & 15;
  int mtile = blockIdx.y * 128 + wave * 16;
  int nbase = blockIdx.x * 128;

  // cooperative async stage of one 128x32 B tile: 512 x 16B chunks
  auto issueB = [&](int buf, int k0) {
    for (int i = 0; i < 2; ++i) {
      int chunk = tid + 256 * i;
      int col = chunk >> 2, part = chunk & 3;
      const unsigned short* gp =
          xt + (size_t)(nbase + col) * C_ + k0 + part * 8;
      unsigned lds = (unsigned)(uintptr_t)&tileB[buf][col][part * 8];
      async_g2l_b128(lds, gp);
    }
  };

  const unsigned short* abase = wqg + (size_t)mtile * C_;
  v8f acc[8] = {};
  issueB(0, 0);
  for (int ks = 0; ks < 16; ++ks) {
    int k0 = ks * 32;
    wait_asynccnt0();       // own async loads done
    __syncthreads();        // whole tile visible to all waves
    if (ks + 1 < 16) issueB((ks + 1) & 1, k0 + 32);
    int buf = ks & 1;
    v16bf a = load_frag_a(abase, C_, k0);
    for (int t = 0; t < 8; ++t) {
      v16bf bfr = load_frag_b_lds(&tileB[buf][t * 16][0]);
      acc[t] = wmma_bf16(a, bfr, acc[t]);
    }
  }
  for (int t = 0; t < 8; ++t) {
    int j = nbase + t * 16 + nn;
    float rs = rstd[j], tm = tmu[j];
    int bb = j >> 13, n = j & (N_ - 1);
    for (int r = 0; r < 8; ++r) {
      int o = mtile + r + 8 * grp;
      float val = rs * acc[t][r] - tm * wsum[o] + wb[o];
      if (o < HID) {
        int h = o >> 5, d = o & 31;
        qb[(((size_t)(bb * H_ + h) * N_) + n) * DH + d] = f2bf(val * SCALE_);
      } else if (o < 2 * HID) {
        kb[((size_t)bb * HID + (o - HID)) * N_ + n] = f2bf(val);
      } else {
        vb[((size_t)bb * HID + (o - 2 * HID)) * N_ + n] = f2bf(val);
      }
    }
  }
}

// ---------------------------------------------------------------------------
// K3: softmax over n (8192) for each of the 2048 rows of k, in place (k is
//     rewritten by K2 every launch, so this is replay-safe).
// ---------------------------------------------------------------------------
__global__ void k3_softmax(unsigned short* __restrict__ kb) {
  __shared__ float red[256];
  int tid = threadIdx.x;
  unsigned short* row = kb + (size_t)blockIdx.x * N_;

  float mx = -3.0e38f;
  for (int i = tid; i < N_; i += 256) mx = fmaxf(mx, bf2f(row[i]));
  red[tid] = mx; __syncthreads();
  for (int s = 128; s > 0; s >>= 1) {
    if (tid < s) red[tid] = fmaxf(red[tid], red[tid + s]);
    __syncthreads();
  }
  mx = red[0]; __syncthreads();

  float sum = 0.f;
  for (int i = tid; i < N_; i += 256) sum += __expf(bf2f(row[i]) - mx);
  red[tid] = sum; __syncthreads();
  for (int s = 128; s > 0; s >>= 1) {
    if (tid < s) red[tid] += red[tid + s];
    __syncthreads();
  }
  float inv = 1.0f / red[0];
  for (int i = tid; i < N_; i += 256)
    row[i] = f2bf(__expf(bf2f(row[i]) - mx) * inv);
}

// ---------------------------------------------------------------------------
// K4: per-head context[32,32] = k_soft[32,8192] @ v^T[8192,32], bf16 WMMA,
//     4 waves split the K(=n) range, LDS reduce, f32 result ctx[bh][d][e].
// ---------------------------------------------------------------------------
__global__ void k4_context(const unsigned short* __restrict__ kb,
                           const unsigned short* __restrict__ vb,
                           float* __restrict__ ctx) {
  __shared__ float red4[4][2][2][8][32];
  int tid = threadIdx.x, wave = tid >> 5, lane = tid & 31;
  int bh = blockIdx.x, b = bh >> 3, h = bh & 7;
  const unsigned short* kbase = kb + ((size_t)b * HID + h * DH) * N_;
  const unsigned short* vbase = vb + ((size_t)b * HID + h * DH) * N_;

  v8f acc[2][2] = {};
  for (int it = 0; it < 64; ++it) {
    int k0 = wave * 2048 + it * 32;
    v16bf a0 = load_frag_a(kbase, N_, k0);
    v16bf a1 = load_frag_a(kbase + (size_t)16 * N_, N_, k0);
    v16bf b0 = load_frag_b(vbase, N_, k0);
    v16bf b1 = load_frag_b(vbase + (size_t)16 * N_, N_, k0);
    acc[0][0] = wmma_bf16(a0, b0, acc[0][0]);
    acc[0][1] = wmma_bf16(a0, b1, acc[0][1]);
    acc[1][0] = wmma_bf16(a1, b0, acc[1][0]);
    acc[1][1] = wmma_bf16(a1, b1, acc[1][1]);
  }
  for (int mi = 0; mi < 2; ++mi)
    for (int ni = 0; ni < 2; ++ni)
      for (int r = 0; r < 8; ++r)
        red4[wave][mi][ni][r][lane] = acc[mi][ni][r];
  __syncthreads();
  for (int u = 0; u < 8; ++u) {
    int li = tid * 8 + u;                 // 0..1023
    int lane_i = li & 31;
    int r  = (li >> 5) & 7;
    int ni = (li >> 8) & 1;
    int mi = (li >> 9) & 1;
    float s = red4[0][mi][ni][r][lane_i] + red4[1][mi][ni][r][lane_i] +
              red4[2][mi][ni][r][lane_i] + red4[3][mi][ni][r][lane_i];
    int d = mi * 16 + r + 8 * (lane_i >> 4);
    int e = ni * 16 + (lane_i & 15);
    ctx[(size_t)bh * (DH * DH) + d * DH + e] = s;
  }
}

// ---------------------------------------------------------------------------
// K5: attn[b][n][o] = (ctx^T @ q) per head: single K=32 WMMA step per tile.
// ---------------------------------------------------------------------------
__global__ void k5_attn(const float* __restrict__ ctx,
                        const unsigned short* __restrict__ qb,
                        unsigned short* __restrict__ attn) {
  int tid = threadIdx.x, wave = tid >> 5, lane = tid & 31;
  int grp = lane >> 4, nn = lane & 15;
  int b = blockIdx.x >> 6;
  int n0 = (blockIdx.x & 63) * 128;
  int ncol0 = n0 + wave * 16;

  for (int h = 0; h < H_; ++h) {
    const float* cp = ctx + (size_t)(b * H_ + h) * (DH * DH);
    v16bf bfrag = load_frag_b(qb + ((size_t)(b * H_ + h) * N_ + ncol0) * DH, DH, 0);
    for (int mi = 0; mi < 2; ++mi) {
      int mt = mi * 16;
      FragBF afr;
      int e = mt + (lane & 15);
      for (int i = 0; i < 8; ++i) {
        int kk = grp * 8 + i;                     // per-ISA A layout
        afr.s[i]     = f2bf(cp[kk * DH + e]);          // K 0..7 / 8..15
        afr.s[8 + i] = f2bf(cp[(kk + 16) * DH + e]);   // K 16..23 / 24..31
      }
      v8f d = {};
      d = wmma_bf16(afr.v, bfrag, d);
      for (int r = 0; r < 8; ++r) {
        int eo = mt + r + 8 * grp;
        int o = h * DH + eo;
        int n = ncol0 + nn;
        attn[(size_t)(b * N_ + n) * HID + o] = f2bf(d[r]);
      }
    }
  }
}

// ---------------------------------------------------------------------------
// K6: final GEMM  y[512,65536] = Wout(bf16) @ attn(bf16) + b_out + x residual,
//     f32 output.  Same async-LDS-staged 128x128 tiling as K2 (K=256).
// ---------------------------------------------------------------------------
__global__ void k6_out_gemm(const unsigned short* __restrict__ wout_bf,
                            const unsigned short* __restrict__ attn,
                            const float* __restrict__ b_out,
                            const float* __restrict__ x,
                            float* __restrict__ out) {
  __shared__ __align__(16) unsigned short tileB[2][128][32];
  int tid = threadIdx.x;
  int wave = tid >> 5, lane = tid & 31;
  int grp = lane >> 4, nn = lane & 15;
  int mtile = blockIdx.y * 128 + wave * 16;
  int nbase = blockIdx.x * 128;

  auto issueB = [&](int buf, int k0) {
    for (int i = 0; i < 2; ++i) {
      int chunk = tid + 256 * i;
      int col = chunk >> 2, part = chunk & 3;
      const unsigned short* gp =
          attn + (size_t)(nbase + col) * HID + k0 + part * 8;
      unsigned lds = (unsigned)(uintptr_t)&tileB[buf][col][part * 8];
      async_g2l_b128(lds, gp);
    }
  };

  const unsigned short* abase = wout_bf + (size_t)mtile * HID;
  v8f acc[8] = {};
  issueB(0, 0);
  for (int ks = 0; ks < 8; ++ks) {
    int k0 = ks * 32;
    wait_asynccnt0();
    __syncthreads();
    if (ks + 1 < 8) issueB((ks + 1) & 1, k0 + 32);
    int buf = ks & 1;
    v16bf a = load_frag_a(abase, HID, k0);
    for (int t = 0; t < 8; ++t) {
      v16bf bfr = load_frag_b_lds(&tileB[buf][t * 16][0]);
      acc[t] = wmma_bf16(a, bfr, acc[t]);
    }
  }
  for (int t = 0; t < 8; ++t) {
    int j = nbase + t * 16 + nn;
    int bb = j >> 13, n = j & (N_ - 1);
    for (int r = 0; r < 8; ++r) {
      int o = mtile + r + 8 * grp;
      size_t idx = ((size_t)bb * C_ + o) * N_ + n;
      out[idx] = acc[t][r] + b_out[o] + x[idx];
    }
  }
}

// ---------------------------------------------------------------------------
extern "C" void kernel_launch(void* const* d_in, const int* in_sizes, int n_in,
                              void* d_out, int out_size, void* d_ws, size_t ws_size,
                              hipStream_t stream) {
  const float* x     = (const float*)d_in[0];
  const float* g     = (const float*)d_in[1];
  const float* b_ln  = (const float*)d_in[2];
  const float* w_qkv = (const float*)d_in[3];
  const float* w_out = (const float*)d_in[4];
  const float* b_out = (const float*)d_in[5];
  float* out = (float*)d_out;

  char* w = (char*)d_ws;
  size_t off = 0;
  auto carve = [&](size_t bytes) {
    size_t o = off;
    off = (off + bytes + 255) & ~(size_t)255;
    return (void*)(w + o);
  };
  unsigned short* wqg     = (unsigned short*)carve((size_t)O3 * C_ * 2);
  unsigned short* wout_bf = (unsigned short*)carve((size_t)C_ * HID * 2);
  float*          wsum    = (float*)carve((size_t)O3 * 4);
  float*          wb      = (float*)carve((size_t)O3 * 4);
  unsigned short* xt      = (unsigned short*)carve((size_t)BN * C_ * 2);
  float*          rstd    = (float*)carve((size_t)BN * 4);
  float*          tmu     = (float*)carve((size_t)BN * 4);
  unsigned short* qb      = (unsigned short*)carve((size_t)B_ * HID * N_ * 2);
  unsigned short* kb      = (unsigned short*)carve((size_t)B_ * HID * N_ * 2);
  unsigned short* vb      = (unsigned short*)carve((size_t)B_ * HID * N_ * 2);
  float*          ctx     = (float*)carve((size_t)B_ * H_ * DH * DH * 4);
  unsigned short* attn    = (unsigned short*)carve((size_t)BN * HID * 2);
  (void)ws_size; (void)in_sizes; (void)n_in; (void)out_size;

  k0_weights<<<O3 + C_, 256, 0, stream>>>(w_qkv, g, b_ln, w_out,
                                          wqg, wout_bf, wsum, wb);
  k1_ln_transpose<<<B_ * (N_ / 64), 256, 0, stream>>>(x, xt, rstd, tmu);
  k2_qkv_gemm<<<dim3(BN / 128, O3 / 128), 256, 0, stream>>>(
      wqg, xt, rstd, tmu, wsum, wb, qb, kb, vb);
  k3_softmax<<<B_ * HID, 256, 0, stream>>>(kb);
  k4_context<<<B_ * H_, 128, 0, stream>>>(kb, vb, ctx);
  k5_attn<<<B_ * (N_ / 128), 256, 0, stream>>>(ctx, qb, attn);
  k6_out_gemm<<<dim3(BN / 128, C_ / 128), 256, 0, stream>>>(
      wout_bf, attn, b_out, x, out);
}
